// Decoder_37271726194816
// MI455X (gfx1250) — matrix operand order repaired and checked by reading, not compiled, not run on hardware
//
#include <hip/hip_runtime.h>

// ---------------------------------------------------------------------------
// CDNA5 (gfx1250) Bahdanau-attention LSTM decoder.
// GEMMs: v_wmma_f32_16x16x32_bf16 with 2-deep async global->LDS pipeline
// (global_load_async_to_lds_b128 / s_wait_asynccnt 4, triple-buffered LDS),
// pre-transposed bf16 weights so both A and B tiles are contiguous copies.
// Attention streams (keys/memory, ~256MB/step bf16) are the HBM-bound term.
// ---------------------------------------------------------------------------

#define USE_ASYNC_COPY 1

typedef __attribute__((ext_vector_type(16))) __bf16 v16bf;
typedef __attribute__((ext_vector_type(8)))  float  v8f;

#define B_   64
#define S_   1024
#define T_   511
#define V_   1024
#define E_   512
#define U_   1024
#define M_   1024
#define KIN_ 2560   // E + U(attn) + U(h)

__device__ __forceinline__ float sigmf(float x) { return 1.0f / (1.0f + __expf(-x)); }

// Safe fast tanh: t = exp(-2|x|); r = (1-t)/(1+t). Saturates cleanly.
__device__ __forceinline__ float fast_tanh(float x) {
    float t = __expf(-2.0f * fabsf(x));
    float r = (1.0f - t) / (1.0f + t);
    return copysignf(r, x);
}

// Low 32 bits of a flat LDS pointer == LDS byte address (ISA flat->LDS rule).
__device__ __forceinline__ unsigned lds_addr_of(const void* p) {
    return (unsigned)(uintptr_t)p;
}

__device__ __forceinline__ void wait_async0() {
#if USE_ASYNC_COPY
    asm volatile("s_wait_asynccnt 0" ::: "memory");
#endif
}
// Drain all but the newest 4 per-wave async copies (i.e. the stage we just
// issued): certifies the *previous* stage's tile has landed in LDS while the
// newest stage's DMA keeps running across the barrier.
__device__ __forceinline__ void wait_async4() {
#if USE_ASYNC_COPY
    asm volatile("s_wait_asynccnt 4" ::: "memory");
#endif
}

// ---------------------------------------------------------------------------
// bf16 WMMA GEMM:  C[M,N] = A[M,K] * Bt[N,K]^T (+ bias[N])
// A row-major [M][K] bf16; Bt row-major [N][K] bf16 (pre-transposed weights).
// Block = 128 threads = 4 waves; block tile 64x64; wave tile 32x32 =
// 2x2 v_wmma_f32_16x16x32_bf16. Triple-buffered LDS; 2-deep async DMA
// pipeline overlapped with WMMA. Requires M%64==0, N%64==0, K%32==0.
// ---------------------------------------------------------------------------

// Both tiles are 64 rows x 32 bf16 (4KB): straight contiguous copies.
// Each thread issues 4 async b128 copies (2 for A, 2 for B) = 4 ASYNCcnt/wave.
__device__ __forceinline__ void stage_tiles_async(const __bf16* __restrict__ A, int lda,
                                                  const __bf16* __restrict__ Bt, int ldbt,
                                                  int row0, int col0, int kk, int tid,
                                                  __bf16 (*As)[40], __bf16 (*Bs)[40]) {
#if USE_ASYNC_COPY
#pragma unroll
    for (int sweep = 0; sweep < 2; ++sweep) {
        int q  = sweep * 128 + tid;       // 256 x 16B chunks per tile
        int r  = q >> 2;                  // row 0..63
        int ch = (q & 3) * 8;             // half-offset 0,8,16,24
        unsigned la = lds_addr_of(&As[r][ch]);
        const __bf16* ga = A + (size_t)(row0 + r) * lda + kk + ch;
        asm volatile("global_load_async_to_lds_b128 %0, %1, off"
                     :: "v"(la), "v"((unsigned long long)(uintptr_t)ga)
                     : "memory");
        unsigned lb = lds_addr_of(&Bs[r][ch]);
        const __bf16* gb = Bt + (size_t)(col0 + r) * ldbt + kk + ch;
        asm volatile("global_load_async_to_lds_b128 %0, %1, off"
                     :: "v"(lb), "v"((unsigned long long)(uintptr_t)gb)
                     : "memory");
    }
#else
#pragma unroll
    for (int sweep = 0; sweep < 4; ++sweep) {
        int l = (sweep * 128 + tid) * 4;
        int r = l >> 5, c = l & 31;
        *reinterpret_cast<uint2*>(&As[r][c]) =
            *reinterpret_cast<const uint2*>(A + (size_t)(row0 + r) * lda + kk + c);
        *reinterpret_cast<uint2*>(&Bs[r][c]) =
            *reinterpret_cast<const uint2*>(Bt + (size_t)(col0 + r) * ldbt + kk + c);
    }
#endif
}

template <bool HAS_BIAS, bool OUT_F32, bool OUT_BF16>
__global__ __launch_bounds__(128)
void gemm_bf16_wmma(const __bf16* __restrict__ A, int lda,
                    const __bf16* __restrict__ Bt, int ldbt,
                    const float* __restrict__ bias,
                    float* __restrict__ C, int ldc,
                    __bf16* __restrict__ Cb, int ldcb,
                    int K)
{
    __shared__ __bf16 As[3][64][40];   // [buf][m][k]
    __shared__ __bf16 Bs[3][64][40];   // [buf][n][k]

    const int tid   = threadIdx.x;
    const int lane  = tid & 31;
    const int wave  = tid >> 5;
    const int wm    = (wave >> 1) * 32;
    const int wn    = (wave & 1)  * 32;
    const int mlane = lane & 15;
    const int g     = lane >> 4;
    const int row0  = blockIdx.y * 64;
    const int col0  = blockIdx.x * 64;

    v8f acc[2][2];
#pragma unroll
    for (int i = 0; i < 2; ++i)
#pragma unroll
        for (int j = 0; j < 2; ++j)
#pragma unroll
            for (int r = 0; r < 8; ++r) acc[i][j][r] = 0.0f;

    const int nk = K >> 5;

    // prologue: prime two stages of the pipeline
    stage_tiles_async(A, lda, Bt, ldbt, row0, col0, 0, tid, As[0], Bs[0]);
    if (nk > 1) {
        stage_tiles_async(A, lda, Bt, ldbt, row0, col0, 32, tid, As[1], Bs[1]);
        wait_async4();          // tile 0 landed; tile 1 DMA still running
    } else {
        wait_async0();
    }
    __syncthreads();

    int cur = 0;
    for (int i = 0; i < nk; ++i) {
        if (i + 2 < nk) {       // keep the DMA two tiles ahead
            int nxt = cur + 2; if (nxt >= 3) nxt -= 3;
            stage_tiles_async(A, lda, Bt, ldbt, row0, col0, (i + 2) << 5, tid,
                              As[nxt], Bs[nxt]);
        }

        // fragments (A and B: e -> K = e + (e>=8?8:0) + 8g)
        __bf16 (*Asb)[40] = As[cur];
        __bf16 (*Bsb)[40] = Bs[cur];
        v16bf af[2], bfr[2];
#pragma unroll
        for (int mt = 0; mt < 2; ++mt) {
            int r = wm + mt * 16 + mlane;
#pragma unroll
            for (int j = 0; j < 8; ++j) {
                int kb = 2 * j + 8 * g + ((j >= 4) ? 8 : 0);
                af[mt][2 * j]     = Asb[r][kb];
                af[mt][2 * j + 1] = Asb[r][kb + 1];
            }
        }
#pragma unroll
        for (int nt = 0; nt < 2; ++nt) {
            int n = wn + nt * 16 + mlane;
#pragma unroll
            for (int e = 0; e < 16; ++e) bfr[nt][e] = Bsb[n][e + 8 * g + ((e >= 8) ? 8 : 0)];
        }
#pragma unroll
        for (int mt = 0; mt < 2; ++mt)
#pragma unroll
            for (int nt = 0; nt < 2; ++nt)
                acc[mt][nt] = __builtin_amdgcn_wmma_f32_16x16x32_bf16(
                    false, af[mt], false, bfr[nt], (short)0, acc[mt][nt],
                    false, false);

        if (i + 1 < nk) {
            if (i + 2 < nk) wait_async4();   // certify tile i+1 only
            else            wait_async0();   // tail: drain everything
            __syncthreads();
        }
        cur = (cur == 2) ? 0 : cur + 1;
    }

    // epilogue: C/D layout row = r + 8*g, col = mlane
#pragma unroll
    for (int mt = 0; mt < 2; ++mt) {
#pragma unroll
        for (int nt = 0; nt < 2; ++nt) {
            int col = col0 + wn + nt * 16 + mlane;
            float bv = 0.0f;
            if constexpr (HAS_BIAS) bv = bias[col];
#pragma unroll
            for (int r = 0; r < 8; ++r) {
                int row = row0 + wm + mt * 16 + r + 8 * g;
                float val = acc[mt][nt][r];
                if constexpr (HAS_BIAS) val += bv;
                if constexpr (OUT_F32)  C[(size_t)row * ldc + col] = val;
                if constexpr (OUT_BF16) Cb[(size_t)row * ldcb + col] = (__bf16)val;
            }
        }
    }
}

// --------------------------- helper kernels --------------------------------

__global__ void k_cast_bf16(const float* __restrict__ in, __bf16* __restrict__ out, int n) {
    int i = blockIdx.x * 256 + threadIdx.x;
    if (i < n) out[i] = (__bf16)in[i];
}

// out[c][r] = bf16(in[r][c]); in: [R][C] f32, out: [C][R] bf16 (coalesced read)
__global__ void k_cast_transpose(const float* __restrict__ in, __bf16* __restrict__ out,
                                 int R, int C) {
    int i = blockIdx.x * 256 + threadIdx.x;
    if (i >= R * C) return;
    int r = i / C, c = i % C;
    out[(size_t)c * R + r] = (__bf16)in[i];
}

// Wkrt[N=4096][K=2560]: Wkrt[c][r] = (r<1536 ? Wk[r][c] : Wr[r-1536][c])
__global__ void k_build_wkrt(const float* __restrict__ Wk, const float* __restrict__ Wr,
                             __bf16* __restrict__ out) {
    int i = blockIdx.x * 256 + threadIdx.x;
    if (i >= 4 * U_ * KIN_) return;
    int c = i / KIN_, r = i % KIN_;
    float v = (r < E_ + U_) ? Wk[(size_t)r * (4 * U_) + c]
                            : Wr[(size_t)(r - (E_ + U_)) * (4 * U_) + c];
    out[i] = (__bf16)v;
}

__global__ void k_embed(const int* __restrict__ tokens, const float* __restrict__ emb,
                        __bf16* __restrict__ xseq) {
    int i = blockIdx.x * 256 + threadIdx.x;
    if (i >= T_ * B_ * E_) return;
    int e = i % E_, rest = i / E_;
    int b = rest % B_, t = rest / B_;
    int tok = tokens[b * T_ + t];
    xseq[(size_t)t * B_ * E_ + b * E_ + e] = (__bf16)emb[(size_t)tok * E_ + e];
}

__global__ void k_init_state(const float* __restrict__ eh, const float* __restrict__ ec,
                             float* __restrict__ h, float* __restrict__ c,
                             float* __restrict__ attn) {
    int i = blockIdx.x * 256 + threadIdx.x;
    if (i >= B_ * U_) return;
    h[i] = eh[i]; c[i] = ec[i]; attn[i] = 0.0f;
}

__global__ void k_cellin(const __bf16* __restrict__ xt, const float* __restrict__ attn,
                         const float* __restrict__ h, __bf16* __restrict__ out) {
    int i = blockIdx.x * 256 + threadIdx.x;
    if (i >= B_ * KIN_) return;
    int b = i / KIN_, j = i % KIN_;
    __bf16 v;
    if (j < E_)            v = xt[b * E_ + j];
    else if (j < E_ + U_)  v = (__bf16)attn[b * U_ + (j - E_)];
    else                   v = (__bf16)h[b * U_ + (j - E_ - U_)];
    out[i] = v;
}

__global__ void k_lstm(const float* __restrict__ z, float* __restrict__ h,
                       float* __restrict__ c, __bf16* __restrict__ hbf) {
    int i = blockIdx.x * 256 + threadIdx.x;
    if (i >= B_ * U_) return;
    int b = i / U_, u = i % U_;
    const float* zr = z + (size_t)b * 4 * U_;
    float ig = sigmf(zr[u]);
    float fg = sigmf(zr[U_ + u]);
    float gg = tanhf(zr[2 * U_ + u]);
    float og = sigmf(zr[3 * U_ + u]);
    float cn = fg * c[i] + ig * gg;
    float hn = og * tanhf(cn);
    c[i] = cn; h[i] = hn; hbf[i] = (__bf16)hn;
}

// score[b][s] = sum_u v[u]*tanh(q[b][u]+keys[b][s][u]); one wave per s,
// 64-bit key loads (4 x bf16), fast tanh.
__global__ __launch_bounds__(256)
void k_score(const float* __restrict__ q, const __bf16* __restrict__ keys,
             const float* __restrict__ v, float* __restrict__ score) {
    int bid = blockIdx.x;                       // B_ * (S_/8) blocks
    int b = bid >> 7;                           // S_/8 == 128
    int s = ((bid & 127) << 3) + (threadIdx.x >> 5);
    int lane = threadIdx.x & 31;
    const __bf16* krow = keys + ((size_t)b * S_ + s) * U_;
    const float* qrow = q + (size_t)b * U_;
    float p = 0.0f;
    for (int u = lane * 4; u < U_; u += 128) {
        uint2 kd = *reinterpret_cast<const uint2*>(krow + u);
        const __bf16* kp = reinterpret_cast<const __bf16*>(&kd);
        float4 qv = *reinterpret_cast<const float4*>(qrow + u);
        float4 vv = *reinterpret_cast<const float4*>(v + u);
        p += vv.x * fast_tanh(qv.x + (float)kp[0]);
        p += vv.y * fast_tanh(qv.y + (float)kp[1]);
        p += vv.z * fast_tanh(qv.z + (float)kp[2]);
        p += vv.w * fast_tanh(qv.w + (float)kp[3]);
    }
#pragma unroll
    for (int off = 16; off > 0; off >>= 1) p += __shfl_down(p, off, 32);
    if (lane == 0) score[b * S_ + s] = p;
}

__global__ __launch_bounds__(256)
void k_softmax(const float* __restrict__ score, float* __restrict__ alig) {
    __shared__ float red[256];
    int b = blockIdx.x, tid = threadIdx.x;
    float m = -1e30f;
    for (int s = tid; s < S_; s += 256) m = fmaxf(m, score[b * S_ + s]);
    red[tid] = m; __syncthreads();
    for (int off = 128; off > 0; off >>= 1) {
        if (tid < off) red[tid] = fmaxf(red[tid], red[tid + off]);
        __syncthreads();
    }
    m = red[0]; __syncthreads();
    float sum = 0.0f;
    for (int s = tid; s < S_; s += 256) {
        float e = __expf(score[b * S_ + s] - m);
        alig[b * S_ + s] = e;
        sum += e;
    }
    red[tid] = sum; __syncthreads();
    for (int off = 128; off > 0; off >>= 1) {
        if (tid < off) red[tid] += red[tid + off];
        __syncthreads();
    }
    float inv = 1.0f / red[0];
    for (int s = tid; s < S_; s += 256) alig[b * S_ + s] *= inv;
}

// ctx[b][m..m+1] = sum_s alig[b][s]*memory[b][s][m..m+1]; 2 bf16 per load.
__global__ __launch_bounds__(256)
void k_ctx(const float* __restrict__ alig, const __bf16* __restrict__ membf,
           float* __restrict__ ctx) {
    __shared__ float al[256];
    int b = blockIdx.x >> 1;                       // grid = B_ * (M_/512)
    int m = ((blockIdx.x & 1) << 9) + threadIdx.x * 2;
    float accx = 0.0f, accy = 0.0f;
    for (int s0 = 0; s0 < S_; s0 += 256) {
        al[threadIdx.x] = alig[b * S_ + s0 + threadIdx.x];
        __syncthreads();
        const __bf16* basep = membf + ((size_t)b * S_ + s0) * M_ + m;
        if (s0 + 256 < S_)
            __builtin_prefetch(membf + ((size_t)b * S_ + s0 + 256) * M_ + m, 0, 0);
#pragma unroll 4
        for (int i = 0; i < 256; ++i) {
            uint d = *reinterpret_cast<const uint*>(basep + (size_t)i * M_);
            const __bf16* p = reinterpret_cast<const __bf16*>(&d);
            float a = al[i];
            accx += a * (float)p[0];
            accy += a * (float)p[1];
        }
        __syncthreads();
    }
    ctx[b * M_ + m]     = accx;
    ctx[b * M_ + m + 1] = accy;
}

__global__ void k_attnin(const float* __restrict__ h, const float* __restrict__ ctx,
                         __bf16* __restrict__ out) {
    int i = blockIdx.x * 256 + threadIdx.x;
    if (i >= B_ * 2 * U_) return;
    int b = i / (2 * U_), j = i % (2 * U_);
    float v = (j < U_) ? h[b * U_ + j] : ctx[b * U_ + (j - U_)];
    out[i] = (__bf16)v;
}

// --------------------------------------------------------------------------

extern "C" void kernel_launch(void* const* d_in, const int* in_sizes, int n_in,
                              void* d_out, int out_size, void* d_ws, size_t ws_size,
                              hipStream_t stream) {
    (void)in_sizes; (void)n_in; (void)out_size; (void)ws_size;

    const int*   tokens = (const int*)  d_in[0];
    const float* memory = (const float*)d_in[1];
    const float* enc_h  = (const float*)d_in[2];
    const float* enc_c  = (const float*)d_in[3];
    const float* emb    = (const float*)d_in[4];
    const float* Wk     = (const float*)d_in[5];
    const float* Wr     = (const float*)d_in[6];
    const float* bgate  = (const float*)d_in[7];
    const float* Wm     = (const float*)d_in[8];
    const float* Wq     = (const float*)d_in[9];
    const float* vvec   = (const float*)d_in[10];
    const float* Wa     = (const float*)d_in[11];
    const float* Wfc    = (const float*)d_in[12];
    const float* bfc    = (const float*)d_in[13];
    float* out = (float*)d_out;

    // ---- workspace carve-up (256B aligned) ----
    char* base = (char*)d_ws;
    size_t off = 0;
    auto carve = [&](size_t bytes) -> char* {
        off = (off + 255) & ~(size_t)255;
        char* p = base + off;
        off += bytes;
        return p;
    };
    __bf16* membf  = (__bf16*)carve((size_t)B_ * S_ * M_ * 2);
    __bf16* keysbf = (__bf16*)carve((size_t)B_ * S_ * U_ * 2);
    __bf16* Wmt    = (__bf16*)carve((size_t)U_ * M_ * 2);        // [U][M]
    __bf16* Wkrt   = (__bf16*)carve((size_t)4 * U_ * KIN_ * 2);  // [4U][2560]
    __bf16* Wqt    = (__bf16*)carve((size_t)U_ * U_ * 2);        // [U][U]
    __bf16* Wat    = (__bf16*)carve((size_t)U_ * 2 * U_ * 2);    // [U][2U]
    __bf16* Wfct   = (__bf16*)carve((size_t)V_ * U_ * 2);        // [V][U]
    __bf16* xseq   = (__bf16*)carve((size_t)T_ * B_ * E_ * 2);
    float*  hS     = (float*) carve((size_t)B_ * U_ * 4);
    float*  cS     = (float*) carve((size_t)B_ * U_ * 4);
    float*  attnF  = (float*) carve((size_t)B_ * U_ * 4);
    __bf16* hbf    = (__bf16*)carve((size_t)B_ * U_ * 2);
    __bf16* attnBf = (__bf16*)carve((size_t)B_ * U_ * 2);
    __bf16* cellin = (__bf16*)carve((size_t)B_ * KIN_ * 2);
    __bf16* attnin = (__bf16*)carve((size_t)B_ * 2 * U_ * 2);
    float*  zbuf   = (float*) carve((size_t)B_ * 4 * U_ * 4);
    float*  qbuf   = (float*) carve((size_t)B_ * U_ * 4);
    float*  score  = (float*) carve((size_t)B_ * S_ * 4);
    float*  alig   = (float*) carve((size_t)B_ * S_ * 4);
    float*  ctx    = (float*) carve((size_t)B_ * U_ * 4);

    // ---- one-time prep: casts + weight transposes ----
    k_cast_bf16<<<(B_ * S_ * M_ + 255) / 256, 256, 0, stream>>>(memory, membf, B_ * S_ * M_);
    k_cast_transpose<<<(M_ * U_ + 255) / 256, 256, 0, stream>>>(Wm, Wmt, M_, U_);
    k_cast_transpose<<<(U_ * U_ + 255) / 256, 256, 0, stream>>>(Wq, Wqt, U_, U_);
    k_cast_transpose<<<(2 * U_ * U_ + 255) / 256, 256, 0, stream>>>(Wa, Wat, 2 * U_, U_);
    k_cast_transpose<<<(U_ * V_ + 255) / 256, 256, 0, stream>>>(Wfc, Wfct, U_, V_);
    k_build_wkrt<<<(4 * U_ * KIN_ + 255) / 256, 256, 0, stream>>>(Wk, Wr, Wkrt);
    k_embed<<<(T_ * B_ * E_ + 255) / 256, 256, 0, stream>>>(tokens, emb, xseq);
    k_init_state<<<(B_ * U_ + 255) / 256, 256, 0, stream>>>(enc_h, enc_c, hS, cS, attnF);

    // keys = memory @ Wm  -> bf16 [B*S, U]
    {
        dim3 grid(U_ / 64, (B_ * S_) / 64);
        gemm_bf16_wmma<false, false, true><<<grid, 128, 0, stream>>>(
            membf, M_, Wmt, M_, nullptr, nullptr, 0, keysbf, U_, M_);
    }

    // ---- sequential decode ----
    for (int t = 0; t < T_; ++t) {
        const __bf16* xt = xseq + (size_t)t * B_ * E_;

        k_cellin<<<(B_ * KIN_ + 255) / 256, 256, 0, stream>>>(xt, attnF, hS, cellin);

        // z = cellin @ Wkr + b   [64 x 4096]
        {
            dim3 grid(4 * U_ / 64, B_ / 64);
            gemm_bf16_wmma<true, true, false><<<grid, 128, 0, stream>>>(
                cellin, KIN_, Wkrt, KIN_, bgate, zbuf, 4 * U_, nullptr, 0, KIN_);
        }

        k_lstm<<<(B_ * U_ + 255) / 256, 256, 0, stream>>>(zbuf, hS, cS, hbf);

        // q = h_new @ Wq   [64 x 1024]
        {
            dim3 grid(U_ / 64, B_ / 64);
            gemm_bf16_wmma<false, true, false><<<grid, 128, 0, stream>>>(
                hbf, U_, Wqt, U_, nullptr, qbuf, U_, nullptr, 0, U_);
        }

        k_score<<<B_ * (S_ / 8), 256, 0, stream>>>(qbuf, keysbf, vvec, score);
        k_softmax<<<B_, 256, 0, stream>>>(score, alig);
        k_ctx<<<B_ * (M_ / 512), 256, 0, stream>>>(alig, membf, ctx);

        k_attnin<<<(B_ * 2 * U_ + 255) / 256, 256, 0, stream>>>(hS, ctx, attnin);

        // attn = [h, ctx] @ Wa   [64 x 1024], f32 + bf16 outputs
        {
            dim3 grid(U_ / 64, B_ / 64);
            gemm_bf16_wmma<false, true, true><<<grid, 128, 0, stream>>>(
                attnin, 2 * U_, Wat, 2 * U_, nullptr, attnF, U_, attnBf, U_, 2 * U_);
        }

        // out[:, t, :] = attn @ Wfc + bfc
        {
            dim3 grid(V_ / 64, B_ / 64);
            gemm_bf16_wmma<true, true, false><<<grid, 128, 0, stream>>>(
                attnBf, U_, Wfct, U_, bfc, out + (size_t)t * V_, T_ * V_,
                nullptr, 0, U_);
        }
    }
}
